// HiddenGCNEncoder_3513283248864
// MI455X (gfx1250) — matrix operand order repaired and checked by reading, not compiled, not run on hardware
//
#include <hip/hip_runtime.h>
#include <hip/hip_bf16.h>

typedef __attribute__((ext_vector_type(16))) __bf16 bf16x16;
typedef __attribute__((ext_vector_type(8)))  float  f32x8;

#define F_IN 256
#define H1   128
#define H2   64

// ---------------- init kernels ----------------
__global__ void k_init_deg(float* __restrict__ deg, int n) {
    int i = blockIdx.x * blockDim.x + threadIdx.x;
    if (i < n) deg[i] = 1.0f;               // self-loop contributes 1 to in-degree
}

__global__ void k_init_bias128(float* __restrict__ a, const float* __restrict__ b, long long total) {
    long long i = (long long)blockIdx.x * blockDim.x + threadIdx.x;
    if (i < total) a[i] = b[i & (H1 - 1)];  // agg1 pre-seeded with b1
}

__global__ void k_init_out(float* __restrict__ out_mu, float* __restrict__ out_lv,
                           const float* __restrict__ bmu, const float* __restrict__ blv,
                           long long total) {
    long long i = (long long)blockIdx.x * blockDim.x + threadIdx.x;
    if (i < total) {
        int c = (int)(i & (H2 - 1));
        out_mu[i] = bmu[c];
        out_lv[i] = blv[c];
    }
}

// ---------------- degree / norm ----------------
__global__ void k_degree(const int* __restrict__ dst, float* __restrict__ deg, int E) {
    int e = blockIdx.x * blockDim.x + threadIdx.x;
    if (e < E) atomicAdd(&deg[dst[e]], 1.0f);
}

__global__ void k_dinv(const float* __restrict__ deg, float* __restrict__ dinv, int n) {
    int i = blockIdx.x * blockDim.x + threadIdx.x;
    if (i < n) dinv[i] = rsqrtf(deg[i]);    // deg >= 1 always (self-loops)
}

// ---------------- weight repack into WMMA B-fragment order (bf16) ----------------
// P[((kt*8 + ct)*32 + lane)*16 + e] = W[kt*32 + (lane>=16?8:0) + (e&7) + (e>=8?16:0)][ct*16 + (lane&15)]
__global__ void k_pack_w1(const float* __restrict__ W, __bf16* __restrict__ P) {
    int idx = blockIdx.x * blockDim.x + threadIdx.x;   // 8 kt * 8 ct * 32 * 16 = 32768
    if (idx >= 8 * 8 * 32 * 16) return;
    int e    = idx & 15;
    int lane = (idx >> 4) & 31;
    int ct   = (idx >> 9) & 7;
    int kt   = (idx >> 12) & 7;
    int k = kt * 32 + ((lane & 16) ? 8 : 0) + (e & 7) + ((e & 8) ? 16 : 0);
    int c = ct * 16 + (lane & 15);
    P[idx] = (__bf16)W[k * H1 + c];
}

// packed [Wmu | Wlv] -> 128x128 virtual matrix, 4 k-tiles
__global__ void k_pack_w2(const float* __restrict__ Wmu, const float* __restrict__ Wlv,
                          __bf16* __restrict__ P) {
    int idx = blockIdx.x * blockDim.x + threadIdx.x;   // 4 kt * 8 ct * 32 * 16 = 16384
    if (idx >= 4 * 8 * 32 * 16) return;
    int e    = idx & 15;
    int lane = (idx >> 4) & 31;
    int ct   = (idx >> 9) & 7;
    int kt   = (idx >> 12) & 3;
    int k = kt * 32 + ((lane & 16) ? 8 : 0) + (e & 7) + ((e & 8) ? 16 : 0);
    int c = ct * 16 + (lane & 15);
    P[idx] = (__bf16)((c < H2) ? Wmu[k * H2 + c] : Wlv[k * H2 + (c - H2)]);
}

// ---------------- GEMM1: T[N,128] = X[N,256] @ W1[256,128] (bf16 WMMA, f32 acc) ----------------
__global__ __launch_bounds__(256) void k_gemm1(const float* __restrict__ X,
                                               const bf16x16* __restrict__ BP,
                                               float* __restrict__ T, int nrows) {
    const int wave = threadIdx.x >> 5;
    const int lane = threadIdx.x & 31;
    const bool hi  = lane >= 16;
    const int m0   = blockIdx.x * 16;
    const int col  = wave * 16 + (lane & 15);   // 8 waves cover 128 cols
    int rowA = m0 + (lane & 15);
    if (rowA >= nrows) rowA = nrows - 1;        // keep EXEC all-ones

    f32x8 acc = {};
    const float* xrow = X + (size_t)rowA * F_IN;
    #pragma unroll
    for (int k0 = 0; k0 < F_IN; k0 += 32) {
        bf16x16 a;
        const float* xp = xrow + k0 + (hi ? 8 : 0);
        #pragma unroll
        for (int e = 0; e < 8; ++e) {
            a[e]     = (__bf16)xp[e];           // K = base+0..7
            a[e + 8] = (__bf16)xp[e + 16];      // K = base+16..23
        }
        bf16x16 b = BP[(((k0 >> 5) * 8 + wave) * 32) + lane];  // pre-packed fragment
        acc = __builtin_amdgcn_wmma_f32_16x16x32_bf16(false, a, false, b,
                                                      (short)0, acc, false, false);
    }
    const int rbase = m0 + (hi ? 8 : 0);
    #pragma unroll
    for (int r = 0; r < 8; ++r) {
        int row = rbase + r;
        if (row < nrows) T[(size_t)row * H1 + col] = acc[r];
    }
}

// ---------------- GEMM2: T[N,128] = relu(A[N,128]) @ packed[Wmu|Wlv] ----------------
__global__ __launch_bounds__(256) void k_gemm2(const float* __restrict__ A,
                                               const bf16x16* __restrict__ BP,
                                               float* __restrict__ T, int nrows) {
    const int wave = threadIdx.x >> 5;
    const int lane = threadIdx.x & 31;
    const bool hi  = lane >= 16;
    const int m0   = blockIdx.x * 16;
    const int col  = wave * 16 + (lane & 15);
    int rowA = m0 + (lane & 15);
    if (rowA >= nrows) rowA = nrows - 1;

    f32x8 acc = {};
    const float* arow = A + (size_t)rowA * H1;
    #pragma unroll
    for (int k0 = 0; k0 < H1; k0 += 32) {
        bf16x16 a;
        const float* ap = arow + k0 + (hi ? 8 : 0);
        #pragma unroll
        for (int e = 0; e < 8; ++e) {
            a[e]     = (__bf16)fmaxf(ap[e], 0.0f);       // fused ReLU
            a[e + 8] = (__bf16)fmaxf(ap[e + 16], 0.0f);
        }
        bf16x16 b = BP[(((k0 >> 5) * 8 + wave) * 32) + lane];
        acc = __builtin_amdgcn_wmma_f32_16x16x32_bf16(false, a, false, b,
                                                      (short)0, acc, false, false);
    }
    const int rbase = m0 + (hi ? 8 : 0);
    #pragma unroll
    for (int r = 0; r < 8; ++r) {
        int row = rbase + r;
        if (row < nrows) T[(size_t)row * H1 + col] = acc[r];
    }
}

// ---------------- edge aggregation: one wave per edge, 128 features ----------------
// split==0: out0[d*128 + c] += T[s*128+c]*w   (c = 0..127)
// split==1: c<64 -> out0[d*64+c], c>=64 -> out1[d*64+c-64]
__global__ __launch_bounds__(256) void k_aggregate(const float* __restrict__ T,
                                                   const int* __restrict__ src,
                                                   const int* __restrict__ dst,
                                                   const float* __restrict__ dinv,
                                                   float* __restrict__ out0,
                                                   float* __restrict__ out1,
                                                   int split, int E_, int N_) {
    const int lane = threadIdx.x & 31;
    const size_t waveId = (size_t)blockIdx.x * (blockDim.x >> 5) + (threadIdx.x >> 5);
    const size_t nWaves = (size_t)gridDim.x * (blockDim.x >> 5);
    const size_t total  = (size_t)E_ + (size_t)N_;

    for (size_t e = waveId; e < total; e += nWaves) {
        if (e + nWaves < (size_t)E_) {                 // prefetch next edge pair
            __builtin_prefetch(&src[e + nWaves], 0, 0);
            __builtin_prefetch(&dst[e + nWaves], 0, 0);
        }
        int s, d;
        if (e < (size_t)E_) { s = src[e]; d = dst[e]; }
        else                { s = d = (int)(e - (size_t)E_); }   // self-loop
        const float w = dinv[s] * dinv[d];
        const float* t = T + (size_t)s * H1;
        if (split == 0) {
            float* o = out0 + (size_t)d * H1;
            #pragma unroll
            for (int j = 0; j < 4; ++j) {
                int c = lane + j * 32;
                atomicAdd(&o[c], t[c] * w);
            }
        } else {
            float* om = out0 + (size_t)d * H2;
            float* ol = out1 + (size_t)d * H2;
            #pragma unroll
            for (int j = 0; j < 2; ++j) {
                int c = lane + j * 32;
                atomicAdd(&om[c], t[c] * w);           // mu half (c < 64)
            }
            #pragma unroll
            for (int j = 2; j < 4; ++j) {
                int c = lane + j * 32;
                atomicAdd(&ol[c - H2], t[c] * w);      // logvar half
            }
        }
    }
}

extern "C" void kernel_launch(void* const* d_in, const int* in_sizes, int n_in,
                              void* d_out, int out_size, void* d_ws, size_t ws_size,
                              hipStream_t stream) {
    const float* x    = (const float*)d_in[0];
    const int*   ei   = (const int*)  d_in[1];
    const float* W1   = (const float*)d_in[2];
    const float* b1   = (const float*)d_in[3];
    const float* Wmu  = (const float*)d_in[4];
    const float* bmu  = (const float*)d_in[5];
    const float* Wlv  = (const float*)d_in[6];
    const float* blv  = (const float*)d_in[7];

    const int N = in_sizes[0] / F_IN;
    const int E = in_sizes[1] / 2;
    const int* src = ei;
    const int* dst = ei + E;

    // workspace layout (floats): deg[N] | dinv[N] | t[N*128] | agg1[N*128] | packed weights
    float* ws   = (float*)d_ws;
    float* deg  = ws;
    float* dinv = deg + N;
    float* t    = dinv + N;                 // reused for t1 and t2
    float* agg1 = t + (size_t)N * H1;
    __bf16* wp1 = (__bf16*)(agg1 + (size_t)N * H1);   // 32768 bf16 (64KB), 32B-aligned
    __bf16* wp2 = wp1 + 8 * 8 * 32 * 16;              // 16384 bf16 (32KB)

    float* out_mu = (float*)d_out;
    float* out_lv = out_mu + (size_t)N * H2;

    const int TB = 256;
    const long long agg1Elems = (long long)N * H1;
    const long long outElems  = (long long)N * H2;

    // 1) init + weight repack (weights only depend on inputs)
    k_init_deg<<<(N + TB - 1) / TB, TB, 0, stream>>>(deg, N);
    k_init_bias128<<<(int)((agg1Elems + TB - 1) / TB), TB, 0, stream>>>(agg1, b1, agg1Elems);
    k_init_out<<<(int)((outElems + TB - 1) / TB), TB, 0, stream>>>(out_mu, out_lv, bmu, blv, outElems);
    k_pack_w1<<<(8 * 8 * 32 * 16) / TB, TB, 0, stream>>>(W1, wp1);
    k_pack_w2<<<(4 * 8 * 32 * 16) / TB, TB, 0, stream>>>(Wmu, Wlv, wp2);

    // 2) degrees + symmetric norm
    k_degree<<<(E + TB - 1) / TB, TB, 0, stream>>>(dst, deg, E);
    k_dinv<<<(N + TB - 1) / TB, TB, 0, stream>>>(deg, dinv, N);

    // 3) t1 = X @ W1  (WMMA bf16, fragment-packed B)
    const int mblocks = (N + 15) / 16;
    k_gemm1<<<mblocks, TB, 0, stream>>>(x, (const bf16x16*)wp1, t, N);

    // 4) agg1 += scatter(norm * t1[src])   (agg1 pre-seeded with b1)
    const int aggBlocks = 2048;             // 16384 waves
    k_aggregate<<<aggBlocks, TB, 0, stream>>>(t, src, dst, dinv, agg1, nullptr, 0, E, N);

    // 5) t2 = relu(agg1) @ packed[Wmu|Wlv]  (reuses t buffer)
    k_gemm2<<<mblocks, TB, 0, stream>>>(agg1, (const bf16x16*)wp2, t, N);

    // 6) scatter into (mu, logvar) halves of d_out (pre-seeded with biases)
    k_aggregate<<<aggBlocks, TB, 0, stream>>>(t, src, dst, dinv, out_mu, out_lv, 1, E, N);
}
